// Decoder_15951508538168
// MI455X (gfx1250) — matrix operand order repaired and checked by reading, not compiled, not run on hardware
//
#include <hip/hip_runtime.h>

typedef _Float16 f16;
typedef __attribute__((ext_vector_type(16))) _Float16 v16h;
typedef __attribute__((ext_vector_type(8)))  float    v8f;

#define B_    2048
#define T_    3
#define N_    24
#define INF_  64
#define FEAT_ 128
#define HID_  64
#define OUT_  64
#define RIN_  192
#define H3_   192
#define PH_   16
#define TLDS_ROWS 768
#define TLDS_LD   40   // row pitch in halves: 80B keeps 16B alignment, bank-friendly

union F16x16 { v16h v; uint4 q[2]; };

__device__ inline v8f wmma_f16(v16h a, v16h b, v8f c) {
  // v_wmma_f32_16x16x32_f16: D = A(16x32) * B(32x16) + C
  return __builtin_amdgcn_wmma_f32_16x16x32_f16(false, a, false, b, (short)0, c,
                                                false, false);
}

// A fragment (16M x 32K, f16): per ISA table, lane<16 holds M=lane, K {0..7,16..23};
// lane>=16 holds M=lane-16, K {8..15,24..31}. Rows at `base` with row stride `ld` halves.
__device__ inline v16h load_afrag(const f16* base, int ld) {
  const int lane = threadIdx.x & 31;
  const int m  = lane & 15;
  const int k0 = (lane < 16) ? 0 : 8;
  const f16* p = base + (size_t)m * ld + k0;
  F16x16 u;
  u.q[0] = *(const uint4*)(p);
  u.q[1] = *(const uint4*)(p + 16);
  return u.v;
}

// B fragment (32K x 16N, f16): lane<16 -> col=lane, K {0..7,16..23};
// lane>=16 -> col=lane-16, K {8..15,24..31}. Column-major source, col stride `ldk`.
__device__ inline v16h load_bfrag(const f16* base, int ldk) {
  const int lane = threadIdx.x & 31;
  const int c  = lane & 15;
  const int k0 = (lane < 16) ? 0 : 8;
  const f16* p = base + (size_t)c * ldk + k0;
  F16x16 u;
  u.q[0] = *(const uint4*)(p);
  u.q[1] = *(const uint4*)(p + 16);
  return u.v;
}

// ---------------- prep kernels ----------------

// L1-row-normalize the four 24x24 G matrices into f16, zero-padded to 32x32 each.
__global__ void k_prep_g(const float* g0, const float* g1, const float* g2,
                         const float* g3, f16* gn) {
  int t = threadIdx.x;
  if (t >= 128) return;
  int g = t >> 5, r = t & 31;
  const float* G = (g == 0) ? g0 : (g == 1) ? g1 : (g == 2) ? g2 : g3;
  f16* dst = gn + g * 1024 + r * 32;
  if (r < N_) {
    float s = 0.f;
    for (int m = 0; m < N_; ++m) s += fabsf(G[r * N_ + m]);
    float inv = 1.f / s;
    for (int m = 0; m < 32; ++m)
      dst[m] = (m < N_) ? (f16)(G[r * N_ + m] * inv) : (f16)0.f;
  } else {
    for (int m = 0; m < 32; ++m) dst[m] = (f16)0.f;
  }
}

// Convert W [n][k][o] fp32 -> Wt [n][o][k] f16 (k contiguous for B fragments).
__global__ void k_prep_w(const float* __restrict__ in, f16* __restrict__ out,
                         int K, int O, int total) {
  int i = blockIdx.x * 256 + threadIdx.x;
  if (i >= total) return;
  int o = i % O; int t = i / O; int k = t % K; int n = t / K;
  out[((size_t)n * O + o) * K + k] = (f16)in[i];
}

// recA = f16 concat(x[:, -1], h); recB = f16 concat(x[:, -2], h); also copy x_t_s out.
__global__ void k_prep_rec(const float* __restrict__ x, const float* __restrict__ h,
                           f16* __restrict__ recA, f16* __restrict__ recB,
                           float* __restrict__ xts, int total) {
  int i = blockIdx.x * 256 + threadIdx.x;
  if (i >= total) return;
  int k = i % RIN_; int t = i / RIN_; int n = t % N_; int b = t / N_;
  float va, vb;
  if (k < INF_) {
    va = x[(((size_t)b * T_ + 2) * N_ + n) * INF_ + k];
    vb = x[(((size_t)b * T_ + 1) * N_ + n) * INF_ + k];
    xts[((size_t)b * N_ + n) * INF_ + k] = va;
  } else {
    float hv = h[((size_t)b * N_ + n) * FEAT_ + (k - INF_)];
    va = hv; vb = hv;
  }
  recA[i] = (f16)va;
  recB[i] = (f16)vb;
}

// ---------------- phase 1: gi and h0 ----------------

__global__ __launch_bounds__(256) void k_init(
    const f16* __restrict__ recA, const f16* __restrict__ recB,
    const f16* __restrict__ WtIH, const f16* __restrict__ WtInit,
    const f16* __restrict__ gn,
    const float* __restrict__ b_ih, const float* __restrict__ b_init,
    float* __restrict__ gi, f16* __restrict__ h0) {
  extern __shared__ __align__(16) char smem[];
  f16* t_lds = (f16*)smem;  // [768][40]
  const int tid = threadIdx.x, wave = tid >> 5, lane = tid & 31;
  const int jj = lane & 15, hb = (lane < 16) ? 0 : 8;
  const int b0 = blockIdx.x * 16;

  for (int i = tid; i < TLDS_ROWS * TLDS_LD; i += 256) t_lds[i] = (f16)0.f;

  v16h gIH[2], gIN[2];
  gIH[0] = load_afrag(gn + 1024, 32);       gIH[1] = load_afrag(gn + 1024 + 512, 32);
  gIN[0] = load_afrag(gn, 32);              gIN[1] = load_afrag(gn + 512, 32);
  __syncthreads();

  // gi = Gih-mix(recA @ W_ih) + b_ih : 192 cols in 4 chunks of 48 (3 parts x 16)
  for (int hc = 0; hc < 4; ++hc) {
    for (int tix = wave; tix < 72; tix += 8) {
      int n = tix / 3, p = tix % 3;
      int o16 = hc * 48 + p * 16;
      const f16* ab = recA + ((size_t)b0 * N_ + n) * RIN_;
      const f16* wb = WtIH + ((size_t)n * H3_ + o16) * RIN_;
      v8f acc = {};
      #pragma unroll
      for (int kt = 0; kt < 6; ++kt)
        acc = wmma_f16(load_afrag(ab + kt * 32, N_ * RIN_),
                       load_bfrag(wb + kt * 32, RIN_), acc);
      #pragma unroll
      for (int j = 0; j < 8; ++j) {
        int b = j + hb;
        t_lds[(b * 48 + p * 16 + jj) * TLDS_LD + n] = (f16)acc[j];
      }
    }
    __syncthreads();
    for (int g = wave; g < 96; g += 8) {
      int mt = g / 48, ct = g % 48;
      v8f z = {};
      v8f d = wmma_f16(gIH[mt], load_bfrag(t_lds + ct * 16 * TLDS_LD, TLDS_LD), z);
      int b = ct / 3, p = ct % 3;
      int o = hc * 48 + p * 16 + jj;
      #pragma unroll
      for (int j = 0; j < 8; ++j) {
        int n = mt * 16 + j + hb;
        if (n < N_)
          gi[((size_t)(b0 + b) * N_ + n) * H3_ + o] = d[j] + b_ih[n * H3_ + o];
      }
    }
    __syncthreads();
  }

  // h0 = Ginit-mix(recB @ W_init) + b_init : 64 cols in 2 chunks of 32
  for (int oc = 0; oc < 2; ++oc) {
    for (int tix = wave; tix < 48; tix += 8) {
      int n = tix >> 1, ot = tix & 1;
      int o16 = oc * 32 + ot * 16;
      const f16* ab = recB + ((size_t)b0 * N_ + n) * RIN_;
      const f16* wb = WtInit + ((size_t)n * HID_ + o16) * RIN_;
      v8f acc = {};
      #pragma unroll
      for (int kt = 0; kt < 6; ++kt)
        acc = wmma_f16(load_afrag(ab + kt * 32, N_ * RIN_),
                       load_bfrag(wb + kt * 32, RIN_), acc);
      #pragma unroll
      for (int j = 0; j < 8; ++j) {
        int b = j + hb;
        t_lds[(b * 32 + ot * 16 + jj) * TLDS_LD + n] = (f16)acc[j];
      }
    }
    __syncthreads();
    for (int g = wave; g < 64; g += 8) {
      int mt = g >> 5, ct = g & 31;
      v8f z = {};
      v8f d = wmma_f16(gIN[mt], load_bfrag(t_lds + ct * 16 * TLDS_LD, TLDS_LD), z);
      int b = ct >> 1;
      int o = oc * 32 + (ct & 1) * 16 + jj;
      #pragma unroll
      for (int j = 0; j < 8; ++j) {
        int n = mt * 16 + j + hb;
        if (n < N_)
          h0[((size_t)(b0 + b) * N_ + n) * HID_ + o] = (f16)(d[j] + b_init[n * HID_ + o]);
      }
    }
    __syncthreads();
  }
}

// ---------------- phase 2: 16-step GRU recurrence, h resident in LDS ----------------

__global__ __launch_bounds__(256) void k_steps(
    const f16* __restrict__ h0, const float* __restrict__ gi,
    const f16* __restrict__ WtHH, const f16* __restrict__ WtFC,
    const f16* __restrict__ gn,
    const float* __restrict__ b_hh, const float* __restrict__ b_fc,
    float* __restrict__ out) {
  extern __shared__ __align__(16) char smem[];
  f16* hbuf  = (f16*)smem;                                 // [2][N][16][HID] ping-pong
  f16* t_lds = (f16*)(smem + 2 * N_ * 16 * HID_ * sizeof(f16)); // [768][40]
  const int tid = threadIdx.x, wave = tid >> 5, lane = tid & 31;
  const int jj = lane & 15, hb = (lane < 16) ? 0 : 8;
  const int b0 = blockIdx.x * 16;

  for (int i = tid; i < N_ * 16 * HID_; i += 256) {
    int n = i / (16 * HID_); int r = i % (16 * HID_); int b = r / HID_; int k = r % HID_;
    hbuf[i] = h0[((size_t)(b0 + b) * N_ + n) * HID_ + k];
  }
  for (int i = tid; i < TLDS_ROWS * TLDS_LD; i += 256) t_lds[i] = (f16)0.f;

  v16h gHH[2], gFC[2];
  gHH[0] = load_afrag(gn + 2 * 1024, 32); gHH[1] = load_afrag(gn + 2 * 1024 + 512, 32);
  gFC[0] = load_afrag(gn + 3 * 1024, 32); gFC[1] = load_afrag(gn + 3 * 1024 + 512, 32);
  __syncthreads();

  int cur = 0;
  for (int step = 0; step < PH_; ++step) {
    const int nxt = cur ^ 1;
    f16* hc_buf = hbuf + cur * (N_ * 16 * HID_);
    f16* hn_buf = hbuf + nxt * (N_ * 16 * HID_);

    for (int hc4 = 0; hc4 < 4; ++hc4) {
      // per-node GEMM t = h_prev @ W_hh for cols {p*64 + hc4*16 + j}, p=0..2
      for (int tix = wave; tix < 72; tix += 8) {
        int n = tix / 3, p = tix % 3;
        int o16 = p * 64 + hc4 * 16;
        const f16* ab = hc_buf + n * (16 * HID_);
        const f16* wb = WtHH + ((size_t)n * H3_ + o16) * HID_;
        v8f acc = {};
        acc = wmma_f16(load_afrag(ab, HID_),      load_bfrag(wb, HID_),      acc);
        acc = wmma_f16(load_afrag(ab + 32, HID_), load_bfrag(wb + 32, HID_), acc);
        #pragma unroll
        for (int j = 0; j < 8; ++j) {
          int b = j + hb;
          t_lds[(b * 48 + p * 16 + jj) * TLDS_LD + n] = (f16)acc[j];
        }
      }
      __syncthreads();
      // graph mixing (Gn as WMMA A, K=32 covers 24 nodes) + GRU gates in-register
      for (int g = wave; g < 32; g += 8) {
        int mt = g >> 4, b = g & 15;
        v8f z = {};
        v8f d0 = wmma_f16(gHH[mt], load_bfrag(t_lds + (b * 48 +  0) * TLDS_LD, TLDS_LD), z);
        v8f d1 = wmma_f16(gHH[mt], load_bfrag(t_lds + (b * 48 + 16) * TLDS_LD, TLDS_LD), z);
        v8f d2 = wmma_f16(gHH[mt], load_bfrag(t_lds + (b * 48 + 32) * TLDS_LD, TLDS_LD), z);
        int o = hc4 * 16 + jj;
        size_t gb = (size_t)(b0 + b) * N_;
        #pragma unroll
        for (int j = 0; j < 8; ++j) {
          int n = mt * 16 + j + hb;
          if (n < N_) {
            size_t gidx = (gb + n) * H3_ + o;
            float ir  = gi[gidx];
            float iu  = gi[gidx + 64];
            float inn = gi[gidx + 128];
            float hr = d0[j] + b_hh[n * H3_ + o];
            float hu = d1[j] + b_hh[n * H3_ + 64 + o];
            float hn = d2[j] + b_hh[n * H3_ + 128 + o];
            float r = 1.f / (1.f + __expf(-(ir + hr)));
            float u = 1.f / (1.f + __expf(-(iu + hu)));
            float nn = tanhf(inn + r * hn);
            float hp = (float)hc_buf[n * (16 * HID_) + b * HID_ + o];
            hn_buf[n * (16 * HID_) + b * HID_ + o] = (f16)((1.f - u) * nn + u * hp);
          }
        }
      }
      __syncthreads();
    }

    // y = tanh(Gfc-mix(h_new @ W_fc) + b_fc), 2 chunks of 32 cols
    for (int oc = 0; oc < 2; ++oc) {
      for (int tix = wave; tix < 48; tix += 8) {
        int n = tix >> 1, ot = tix & 1;
        int o16 = oc * 32 + ot * 16;
        const f16* ab = hn_buf + n * (16 * HID_);
        const f16* wb = WtFC + ((size_t)n * OUT_ + o16) * HID_;
        v8f acc = {};
        acc = wmma_f16(load_afrag(ab, HID_),      load_bfrag(wb, HID_),      acc);
        acc = wmma_f16(load_afrag(ab + 32, HID_), load_bfrag(wb + 32, HID_), acc);
        #pragma unroll
        for (int j = 0; j < 8; ++j) {
          int b = j + hb;
          t_lds[(b * 32 + ot * 16 + jj) * TLDS_LD + n] = (f16)acc[j];
        }
      }
      __syncthreads();
      for (int g = wave; g < 64; g += 8) {
        int mt = g >> 5, ct = g & 31;
        v8f z = {};
        v8f d = wmma_f16(gFC[mt], load_bfrag(t_lds + ct * 16 * TLDS_LD, TLDS_LD), z);
        int b = ct >> 1;
        int o = oc * 32 + (ct & 1) * 16 + jj;
        #pragma unroll
        for (int j = 0; j < 8; ++j) {
          int n = mt * 16 + j + hb;
          if (n < N_)
            out[(((size_t)(b0 + b) * PH_ + step) * N_ + n) * OUT_ + o] =
                tanhf(d[j] + b_fc[n * OUT_ + o]);
        }
      }
      __syncthreads();
    }
    cur = nxt;
  }
}

extern "C" void kernel_launch(void* const* d_in, const int* in_sizes, int n_in,
                              void* d_out, int out_size, void* d_ws, size_t ws_size,
                              hipStream_t stream) {
  (void)in_sizes; (void)n_in; (void)out_size; (void)ws_size;
  const float* x      = (const float*)d_in[0];
  const float* h      = (const float*)d_in[1];
  const float* W_init = (const float*)d_in[4];
  const float* b_init = (const float*)d_in[5];
  const float* G_init = (const float*)d_in[6];
  const float* W_ih   = (const float*)d_in[7];
  const float* b_ih   = (const float*)d_in[8];
  const float* G_ih   = (const float*)d_in[9];
  const float* W_hh   = (const float*)d_in[10];
  const float* b_hh   = (const float*)d_in[11];
  const float* G_hh   = (const float*)d_in[12];
  const float* W_fc   = (const float*)d_in[13];
  const float* b_fc   = (const float*)d_in[14];
  const float* G_fc   = (const float*)d_in[15];

  char* ws = (char*)d_ws;
  size_t off = 0;
  auto alloc = [&](size_t bytes) {
    char* p = ws + off; off += (bytes + 255) & ~(size_t)255; return p;
  };
  f16*   gn     = (f16*)  alloc(4 * 32 * 32 * sizeof(f16));
  f16*   wtInit = (f16*)  alloc((size_t)N_ * HID_ * RIN_ * sizeof(f16));
  f16*   wtIH   = (f16*)  alloc((size_t)N_ * H3_  * RIN_ * sizeof(f16));
  f16*   wtHH   = (f16*)  alloc((size_t)N_ * H3_  * HID_ * sizeof(f16));
  f16*   wtFC   = (f16*)  alloc((size_t)N_ * OUT_ * HID_ * sizeof(f16));
  f16*   recA   = (f16*)  alloc((size_t)B_ * N_ * RIN_ * sizeof(f16));
  f16*   recB   = (f16*)  alloc((size_t)B_ * N_ * RIN_ * sizeof(f16));
  f16*   h0     = (f16*)  alloc((size_t)B_ * N_ * HID_ * sizeof(f16));
  float* gi     = (float*)alloc((size_t)B_ * N_ * H3_ * sizeof(float));

  float* out = (float*)d_out;
  float* xts = out + (size_t)B_ * PH_ * N_ * OUT_;

  k_prep_g<<<1, 128, 0, stream>>>(G_init, G_ih, G_hh, G_fc, gn);
  int tot;
  tot = N_ * RIN_ * HID_; k_prep_w<<<(tot + 255) / 256, 256, 0, stream>>>(W_init, wtInit, RIN_, HID_, tot);
  tot = N_ * RIN_ * H3_;  k_prep_w<<<(tot + 255) / 256, 256, 0, stream>>>(W_ih,   wtIH,   RIN_, H3_,  tot);
  tot = N_ * HID_ * H3_;  k_prep_w<<<(tot + 255) / 256, 256, 0, stream>>>(W_hh,   wtHH,   HID_, H3_,  tot);
  tot = N_ * HID_ * OUT_; k_prep_w<<<(tot + 255) / 256, 256, 0, stream>>>(W_fc,   wtFC,   HID_, OUT_, tot);
  tot = B_ * N_ * RIN_;   k_prep_rec<<<(tot + 255) / 256, 256, 0, stream>>>(x, h, recA, recB, xts, tot);

  size_t smem1 = (size_t)TLDS_ROWS * TLDS_LD * sizeof(f16);
  k_init<<<B_ / 16, 256, smem1, stream>>>(recA, recB, wtIH, wtInit, gn, b_ih, b_init, gi, h0);

  size_t smem2 = (size_t)2 * N_ * 16 * HID_ * sizeof(f16) + smem1;
  k_steps<<<B_ / 16, 256, smem2, stream>>>(h0, gi, wtHH, wtFC, gn, b_hh, b_fc, out);
}
